// ImprovedZAM_Module_17446157156905
// MI455X (gfx1250) — matrix operand order repaired and checked by reading, not compiled, not run on hardware
//
#include <hip/hip_runtime.h>
#include <math.h>

// ---------------------------------------------------------------------------
// Fused attention-gate (ImprovedZAM) for gfx1250 / MI455X.
//   g1 = Wg*g, x1 = Wx*x  (channel GEMMs, bf16 WMMA, f32 accum, K=512 fused)
//   psi = sigmoid(Wpsi . relu(g1+x1)) * S_zernike
//   out = x * psi
// One block = (batch b, 64 spatial positions). g/x each read from HBM once.
// ---------------------------------------------------------------------------

#define BATCH 8
#define FG    256
#define FL    256
#define FINT  128
#define HW    16384          // 128*128
#define NT    64             // spatial tile
#define BST   264            // LDS column stride in bf16 elems (256 + 8 pad)

typedef __bf16 v16bf __attribute__((ext_vector_type(16)));
typedef __bf16 v2bf  __attribute__((ext_vector_type(2)));
typedef float  v2f   __attribute__((ext_vector_type(2)));
typedef float  v8f   __attribute__((ext_vector_type(8)));

// Packed f32x2 -> bf16x2 (RNE); ext-vector convert lowers to hw packed cvt.
__device__ __forceinline__ unsigned pkbf(float lo, float hi) {
  v2bf p = __builtin_convertvector((v2f){lo, hi}, v2bf);
  return __builtin_bit_cast(unsigned, p);
}

__global__ __launch_bounds__(256, 1)
void zam_fused(const float* __restrict__ g,  const float* __restrict__ x,
               const float* __restrict__ Wg, const float* __restrict__ Wx,
               const float* __restrict__ Wpsi, float* __restrict__ out,
               float S) {
  __shared__ unsigned short lds_bt[NT * BST];   // 33,792 B: B-tile, col-major [n][k]
  __shared__ float psi_acc[NT];                 // per-position gate accumulator

  const int tid  = threadIdx.x;
  const int w    = tid >> 5;          // wave 0..7  -> output rows 16w..16w+15
  const int lane = tid & 31;
  const int lm   = lane & 15;
  const int kh8  = (lane < 16) ? 0 : 16;   // A-frag K sub-offset
  const int kh16 = (lane < 16) ? 0 : 16;   // B-frag K sub-offset
  const int q    = tid >> 6;          // 0..3 (staging / epilogue row group)
  const int n    = tid & 63;          // spatial position within tile

  const int b  = blockIdx.x >> 8;          // 8 batches
  const int s0 = (blockIdx.x & 255) << 6;  // 256 tiles of 64 positions

  const float* gb = g + (size_t)b * FG * HW + s0;
  const float* xb = x + (size_t)b * FL * HW + s0;

  if (tid < NT) psi_acc[tid] = 0.0f;

  const int o = 16 * w + lm;          // output row handled by this lane
  v8f acc[4] = {};                    // four 16x16 f32 D tiles (N sub-tiles)

  // ---- K loop in two staged halves (g channels via Wg, then x via Wx) -----
#pragma unroll
  for (int h = 0; h < 2; ++h) {
    // A fragments for this half: 8 K-steps of 16x32 bf16 weights.
    // A 16x32 layout (matches B): lane holds row lm; lanes<16: K {0..15},
    // lanes>=16: K {16..31} (kh8 shifts the run).
    const float* Wm = (h == 0) ? Wg : Wx;
    v16bf afr[8];
#pragma unroll
    for (int ks = 0; ks < 8; ++ks) {
      const float* base = Wm + (size_t)o * 256 + ks * 32 + kh8;
      float4 f0 = *(const float4*)(base + 0);
      float4 f1 = *(const float4*)(base + 4);
      float4 f2 = *(const float4*)(base + 8);
      float4 f3 = *(const float4*)(base + 12);
      union { v16bf v; unsigned u[8]; } A;
      A.u[0] = pkbf(f0.x, f0.y); A.u[1] = pkbf(f0.z, f0.w);
      A.u[2] = pkbf(f1.x, f1.y); A.u[3] = pkbf(f1.z, f1.w);
      A.u[4] = pkbf(f2.x, f2.y); A.u[5] = pkbf(f2.z, f2.w);
      A.u[6] = pkbf(f3.x, f3.y); A.u[7] = pkbf(f3.z, f3.w);
      afr[ks] = A.v;
    }

    // Stage 256 channel rows x 64 positions -> LDS, transposed to [n][k] bf16.
    const float* src = (h == 0) ? gb : xb;
#pragma unroll 4
    for (int j = 0; j < 32; ++j) {
      int k = 2 * (q + 4 * j);
      float v0 = src[(size_t)k * HW + n];
      float v1 = src[(size_t)(k + 1) * HW + n];
      *(unsigned*)&lds_bt[n * BST + k] = pkbf(v0, v1);
    }
    __syncthreads();

#pragma unroll
    for (int ksl = 0; ksl < 8; ++ksl) {
      const int kk = ksl * 32 + kh16;
      // Load all four sub-tile B fragments first (8 back-to-back ds_load_b128),
      // then issue the 4 WMMAs — one dscnt wait per K-step, not four.
      union { v16bf v; uint4 u[2]; } Bf[4];
#pragma unroll
      for (int sub = 0; sub < 4; ++sub) {
        const uint4* p = (const uint4*)&lds_bt[(sub * 16 + lm) * BST + kk];
        Bf[sub].u[0] = p[0];
        Bf[sub].u[1] = p[1];
      }
      v16bf a = afr[ksl];
#pragma unroll
      for (int sub = 0; sub < 4; ++sub) {
        acc[sub] = __builtin_amdgcn_wmma_f32_16x16x32_bf16(
            false, a, false, Bf[sub].v, (short)0, acc[sub], false, false);
      }
    }
    __syncthreads();
  }

  // ---- relu + Wpsi projection: reduce 128 rows -> psi_acc[n] --------------
  // D layout: VGPR r, lanes<16 -> row r, lanes>=16 -> row r+8 (cols = lm).
  const int o0 = 16 * w + ((lane < 16) ? 0 : 8);
  float4 wp0 = *(const float4*)(Wpsi + o0);
  float4 wp1 = *(const float4*)(Wpsi + o0 + 4);
  float wpr[8] = {wp0.x, wp0.y, wp0.z, wp0.w, wp1.x, wp1.y, wp1.z, wp1.w};
#pragma unroll
  for (int sub = 0; sub < 4; ++sub) {
    float s = 0.0f;
#pragma unroll
    for (int r = 0; r < 8; ++r)
      s = fmaf(wpr[r], fmaxf(acc[sub][r], 0.0f), s);
    atomicAdd(&psi_acc[sub * 16 + lm], s);   // ds_add_f32
  }
  __syncthreads();

  if (tid < NT) {
    float v = psi_acc[tid];
    psi_acc[tid] = S / (1.0f + __expf(-v));  // sigmoid * zernike scalar
  }
  __syncthreads();

  // ---- out = x * psi (tile of x is cache-hot; streaming stores) -----------
  const float psiv = psi_acc[n];
  float* ob = out + (size_t)b * FL * HW + s0 + n;
  const float* xr = xb + n;
#pragma unroll 8
  for (int j = 0; j < 64; ++j) {
    int c = q + 4 * j;
    __builtin_nontemporal_store(xr[(size_t)c * HW] * psiv, &ob[(size_t)c * HW]);
  }
}

extern "C" void kernel_launch(void* const* d_in, const int* in_sizes, int n_in,
                              void* d_out, int out_size, void* d_ws, size_t ws_size,
                              hipStream_t stream) {
  const float* g    = (const float*)d_in[0];
  const float* x    = (const float*)d_in[1];
  const float* Wg   = (const float*)d_in[2];
  const float* Wx   = (const float*)d_in[3];
  const float* Wpsi = (const float*)d_in[4];
  float* out = (float*)d_out;

  // Static Zernike scalar: sum cos(q*atan2(j,i)) * (p+1)/(pi*(N-1)^2), p=2,q=1,N=128.
  double s = 0.0;
  for (int i = 0; i < 128; ++i)
    for (int j = 0; j < 128; ++j)
      s += cos(atan2((double)j, (double)i));
  const float S = (float)(s * 3.0 / (M_PI * 127.0 * 127.0));

  dim3 grid(BATCH * (HW / NT));   // 2048 blocks
  dim3 block(256);                // 8 waves (wave32)
  hipLaunchKernelGGL(zam_fused, grid, block, 0, stream, g, x, Wg, Wx, Wpsi, out, S);
}
